// EfficientAttention_35424890258015
// MI455X (gfx1250) — matrix operand order repaired and checked by reading, not compiled, not run on hardware
//
#include <hip/hip_runtime.h>

// EfficientAttention (QANet-style factorized attention), B=4, L=4096, H=1024, nh=16, hc=64.
// Pipeline:
//   1) proj_gemm (WMMA bf16, double-buffered LDS) x3 -> K,Q,V   (x*dw+db) @ pw^T + pb
//   2) key softmax over L (masked, in-place on K)
//   3) query softmax over hc=64 (masked, in-place on Q)
//   4) context_gemm (WMMA): ctx[b,h] = Ksm[64xL] @ V^T[Lx64]
//   5) attend_gemm (WMMA):  agg[b,l,h*64+d] = (ctx^T @ Qsm)    (agg reuses K buffer)
//   6) proj_gemm on agg with r-weights -> d_out
// Workspace: 3*B*L*H + nh*B*64*64 floats ~= 202 MB.

#define L_SEQ 4096
#define H_DIM 1024
#define B_SZ  4
#define NHEAD 16
#define HC    64
#define M_TOT (B_SZ * L_SEQ) /* 16384 */
#define NEG_INF_F (-1.0e30f)

typedef __attribute__((ext_vector_type(16))) __bf16         v16bf;
typedef __attribute__((ext_vector_type(16))) unsigned short v16us;
typedef __attribute__((ext_vector_type(8)))  unsigned short v8us;
typedef __attribute__((ext_vector_type(4)))  unsigned short v4us;
typedef __attribute__((ext_vector_type(8)))  float          v8f;

// f32 -> bf16 via native fptrunc; on gfx1250 this should lower to a hardware
// convert (bf16 is a first-class type here), avoiding the 4-op software RNE.
static __device__ __forceinline__ unsigned short f2bf(float f) {
  __bf16 h = (__bf16)f;
  return __builtin_bit_cast(unsigned short, h);
}

static __device__ __forceinline__ v4us pack4(float a, float b, float c, float d) {
  v4us r;
  r[0] = f2bf(a); r[1] = f2bf(b); r[2] = f2bf(c); r[3] = f2bf(d);
  return r;
}

static __device__ __forceinline__ v16us pack16(float4 q0, float4 q1, float4 q2,
                                               float4 q3) {
  v16us r;
  r[0]  = f2bf(q0.x); r[1]  = f2bf(q0.y); r[2]  = f2bf(q0.z); r[3]  = f2bf(q0.w);
  r[4]  = f2bf(q1.x); r[5]  = f2bf(q1.y); r[6]  = f2bf(q1.z); r[7]  = f2bf(q1.w);
  r[8]  = f2bf(q2.x); r[9]  = f2bf(q2.y); r[10] = f2bf(q2.z); r[11] = f2bf(q2.w);
  r[12] = f2bf(q3.x); r[13] = f2bf(q3.y); r[14] = f2bf(q3.z); r[15] = f2bf(q3.w);
  return r;
}

// Two 16B-aligned LDS reads -> one 16-element bf16 fragment (ds_load_b128 x2).
static __device__ __forceinline__ v16us ld2x8(const unsigned short* p0,
                                              const unsigned short* p1) {
  v8us lo = *(const v8us*)p0;
  v8us hi = *(const v8us*)p1;
  return __builtin_shufflevector(lo, hi, 0, 1, 2, 3, 4, 5, 6, 7, 8, 9, 10, 11,
                                 12, 13, 14, 15);
}

static __device__ __forceinline__ v8f wmma_bf16(v16us au, v16us bu, v8f c) {
  v16bf a = __builtin_bit_cast(v16bf, au);
  v16bf b = __builtin_bit_cast(v16bf, bu);
  return __builtin_amdgcn_wmma_f32_16x16x32_bf16(false, a, false, b, (short)0, c,
                                                 false, false);
}

// ---------------------------------------------------------------------------
// Kernel A: Y[m,n] = sum_k (X[m,k]*dw[k]+db[k]) * pw[n,k] + pb[n]
// Block tile 128(M) x 64(N), 8 waves; wave w: m-tile w, all 4 n-tiles.
// Double-buffered LDS: global loads for tile i+1 issue before WMMAs on tile i.
// ---------------------------------------------------------------------------
__global__ __launch_bounds__(256) void proj_gemm_kernel(
    const float* __restrict__ X, const float* __restrict__ dw,
    const float* __restrict__ db, const float* __restrict__ pw,
    const float* __restrict__ pb, float* __restrict__ Y) {
  // Row stride 40 ushorts = 80 B -> every 8-ushort chunk is 16B aligned.
  __shared__ __align__(16) unsigned short As[2][128][40]; // [m][k]
  __shared__ __align__(16) unsigned short Bs[2][64][40];  // [n][k]

  const int tid  = threadIdx.x;
  const int lane = tid & 31;
  const int wav  = tid >> 5;
  const int n0   = blockIdx.x * 64;
  const int m0   = blockIdx.y * 128;

  const int kq = (tid & 7) * 4; // k quad within 32
  const int rr = tid >> 3;      // 0..31

  float4 xa[4], xb[2];
  float  dq[4], eq[4];

  auto stage_regs = [&](int k0) {
#pragma unroll
    for (int j = 0; j < 4; ++j) {
      dq[j] = dw[k0 + kq + j];
      eq[j] = db[k0 + kq + j];
    }
#pragma unroll
    for (int p = 0; p < 4; ++p)
      xa[p] = *(const float4*)(X + (size_t)(m0 + rr + p * 32) * H_DIM + k0 + kq);
#pragma unroll
    for (int p = 0; p < 2; ++p)
      xb[p] = *(const float4*)(pw + (size_t)(n0 + rr + p * 32) * H_DIM + k0 + kq);
  };
  auto stage_lds = [&](int buf) {
#pragma unroll
    for (int p = 0; p < 4; ++p)
      *(v4us*)&As[buf][rr + p * 32][kq] =
          pack4(fmaf(xa[p].x, dq[0], eq[0]), fmaf(xa[p].y, dq[1], eq[1]),
                fmaf(xa[p].z, dq[2], eq[2]), fmaf(xa[p].w, dq[3], eq[3]));
#pragma unroll
    for (int p = 0; p < 2; ++p)
      *(v4us*)&Bs[buf][rr + p * 32][kq] =
          pack4(xb[p].x, xb[p].y, xb[p].z, xb[p].w);
  };

  v8f acc[4] = {v8f{}, v8f{}, v8f{}, v8f{}};

  stage_regs(0);
  stage_lds(0);

  // Fragment geometry (contiguous k order used here, A and B alike):
  //   A: element e -> K = krow + e, same k order as B
  //   B: element e -> K = krow + e, krow = 0 / 16 per lane half
  const int mrow = wav * 16 + (lane & 15);
  const int kb   = (lane >> 4) * 8;
  const int krow = (lane >> 4) * 16;
  const int nlan = lane & 15;

  const int NIT = H_DIM / 32;
  for (int i = 0; i < NIT; ++i) {
    __syncthreads();
    if (i + 1 < NIT) stage_regs((i + 1) * 32); // issue global loads early
    const int buf = i & 1;
    const v16us au = ld2x8(&As[buf][mrow][krow], &As[buf][mrow][krow + 8]);
    const unsigned short* b0 = &Bs[buf][0 * 16 + nlan][0];
    const unsigned short* b1 = &Bs[buf][1 * 16 + nlan][0];
    const unsigned short* b2 = &Bs[buf][2 * 16 + nlan][0];
    const unsigned short* b3 = &Bs[buf][3 * 16 + nlan][0];
    const v16us bu0 = ld2x8(b0 + krow, b0 + krow + 8);
    const v16us bu1 = ld2x8(b1 + krow, b1 + krow + 8);
    const v16us bu2 = ld2x8(b2 + krow, b2 + krow + 8);
    const v16us bu3 = ld2x8(b3 + krow, b3 + krow + 8);
    acc[0] = wmma_bf16(au, bu0, acc[0]);
    acc[1] = wmma_bf16(au, bu1, acc[1]);
    acc[2] = wmma_bf16(au, bu2, acc[2]);
    acc[3] = wmma_bf16(au, bu3, acc[3]);
    if (i + 1 < NIT) stage_lds((i + 1) & 1); // waits loads, fills other buffer
  }

  // Epilogue: C/D layout -> M = r + 8*(lane/16), N = lane%16
  const int mh = (lane >> 4) * 8;
#pragma unroll
  for (int nt = 0; nt < 4; ++nt) {
    const int n     = n0 + nt * 16 + nlan;
    const float bia = pb[n];
#pragma unroll
    for (int r = 0; r < 8; ++r) {
      const int m = m0 + wav * 16 + mh + r;
      Y[(size_t)m * H_DIM + n] = acc[nt][r] + bia;
    }
  }
}

// ---------------------------------------------------------------------------
// Kernel B: masked softmax over L per (b, channel). In-place on K buffer.
// ---------------------------------------------------------------------------
__global__ __launch_bounds__(256) void key_softmax_kernel(
    float* __restrict__ K, const int* __restrict__ mask) {
  const int b   = blockIdx.x >> 10; // /H_DIM
  const int ch  = blockIdx.x & 1023;
  const int tid = threadIdx.x;
  float* base     = K + (size_t)b * L_SEQ * H_DIM + ch;
  const int* mrow = mask + b * L_SEQ;

  float vals[16];
  float mx = -3.0e38f;
#pragma unroll
  for (int i = 0; i < 16; ++i) {
    const int l = tid + i * 256;
    float v = base[(size_t)l * H_DIM];
    v = (mrow[l] != 0) ? v : NEG_INF_F;
    vals[i] = v;
    mx = fmaxf(mx, v);
  }
  __shared__ float red[256];
  red[tid] = mx;
  __syncthreads();
  for (int s = 128; s > 0; s >>= 1) {
    if (tid < s) red[tid] = fmaxf(red[tid], red[tid + s]);
    __syncthreads();
  }
  mx = red[0];
  __syncthreads();

  float sum = 0.0f;
#pragma unroll
  for (int i = 0; i < 16; ++i) {
    vals[i] = __expf(vals[i] - mx);
    sum += vals[i];
  }
  red[tid] = sum;
  __syncthreads();
  for (int s = 128; s > 0; s >>= 1) {
    if (tid < s) red[tid] += red[tid + s];
    __syncthreads();
  }
  const float inv = 1.0f / red[0];
#pragma unroll
  for (int i = 0; i < 16; ++i) {
    const int l = tid + i * 256;
    base[(size_t)l * H_DIM] = vals[i] * inv;
  }
}

// ---------------------------------------------------------------------------
// Kernel C: masked softmax over hc=64 per (b,l,head). In-place on Q buffer.
// One block per (b,l); 8 waves x 2 heads; wave32 shuffle reductions.
// ---------------------------------------------------------------------------
__global__ __launch_bounds__(256) void query_softmax_kernel(
    float* __restrict__ Q, const int* __restrict__ mask) {
  const int bl   = blockIdx.x; // = b*L + l (matches mask flat index)
  const int lane = threadIdx.x & 31;
  const int wav  = threadIdx.x >> 5;
  float* row    = Q + (size_t)bl * H_DIM;
  const bool on = (mask[bl] != 0);

#pragma unroll
  for (int hh = 0; hh < 2; ++hh) {
    const int h = wav * 2 + hh;
    float v0 = row[h * HC + lane];
    float v1 = row[h * HC + 32 + lane];
    if (!on) { v0 = NEG_INF_F; v1 = NEG_INF_F; }
    float mx = fmaxf(v0, v1);
#pragma unroll
    for (int off = 16; off > 0; off >>= 1) mx = fmaxf(mx, __shfl_xor(mx, off, 32));
    const float e0 = __expf(v0 - mx);
    const float e1 = __expf(v1 - mx);
    float s = e0 + e1;
#pragma unroll
    for (int off = 16; off > 0; off >>= 1) s += __shfl_xor(s, off, 32);
    const float inv = 1.0f / s;
    row[h * HC + lane]      = e0 * inv;
    row[h * HC + 32 + lane] = e1 * inv;
  }
}

// ---------------------------------------------------------------------------
// Kernel D1: ctx[b,h][c][d] = sum_l Ksm[b,l,h*64+c] * V[b,l,h*64+d]
// One block per (b,h); 8 waves cover the 4x4 grid of 16x16 tiles (2 each).
// ---------------------------------------------------------------------------
__global__ __launch_bounds__(256) void context_gemm_kernel(
    const float* __restrict__ Ksm, const float* __restrict__ V,
    float* __restrict__ ctx) {
  const int b = blockIdx.x >> 4;
  const int h = blockIdx.x & 15;
  const float* kbase = Ksm + (size_t)b * L_SEQ * H_DIM + h * HC;
  const float* vbase = V   + (size_t)b * L_SEQ * H_DIM + h * HC;

  __shared__ __align__(16) unsigned short As[64][40]; // [c][l-k]
  __shared__ __align__(16) unsigned short Bs[64][40]; // [d][l-k]

  const int tid   = threadIdx.x;
  const int lane  = tid & 31;
  const int wav   = tid >> 5;
  const int mt    = wav >> 1; // m tile 0..3 (c)
  const int npair = wav & 1;  // n tile pair (d)

  const int lq = tid >> 4;       // l slot 0..15 (two passes -> 32 l)
  const int qc = (tid & 15) * 4; // channel quad

  v8f acc[2] = {v8f{}, v8f{}};

  const int mrow = mt * 16 + (lane & 15);
  const int kb   = (lane >> 4) * 8;
  const int krow = (lane >> 4) * 16;

  for (int l0 = 0; l0 < L_SEQ; l0 += 32) {
    __syncthreads();
#pragma unroll
    for (int p = 0; p < 2; ++p) {
      const int l = lq + p * 16;
      const float4 kv = *(const float4*)(kbase + (size_t)(l0 + l) * H_DIM + qc);
      const float4 vv = *(const float4*)(vbase + (size_t)(l0 + l) * H_DIM + qc);
      As[qc + 0][l] = f2bf(kv.x);
      As[qc + 1][l] = f2bf(kv.y);
      As[qc + 2][l] = f2bf(kv.z);
      As[qc + 3][l] = f2bf(kv.w);
      Bs[qc + 0][l] = f2bf(vv.x);
      Bs[qc + 1][l] = f2bf(vv.y);
      Bs[qc + 2][l] = f2bf(vv.z);
      Bs[qc + 3][l] = f2bf(vv.w);
    }
    __syncthreads();

    const v16us au = ld2x8(&As[mrow][krow], &As[mrow][krow + 8]);
    const unsigned short* b0 = &Bs[(npair * 2 + 0) * 16 + (lane & 15)][0];
    const unsigned short* b1 = &Bs[(npair * 2 + 1) * 16 + (lane & 15)][0];
    const v16us bu0 = ld2x8(b0 + krow, b0 + krow + 8);
    const v16us bu1 = ld2x8(b1 + krow, b1 + krow + 8);
    acc[0] = wmma_bf16(au, bu0, acc[0]);
    acc[1] = wmma_bf16(au, bu1, acc[1]);
  }

  float* cb      = ctx + (size_t)blockIdx.x * (HC * HC);
  const int nlan = lane & 15;
  const int mh   = (lane >> 4) * 8;
#pragma unroll
  for (int j = 0; j < 2; ++j) {
    const int n = (npair * 2 + j) * 16 + nlan;
#pragma unroll
    for (int r = 0; r < 8; ++r) cb[(mt * 16 + mh + r) * HC + n] = acc[j][r];
  }
}

// ---------------------------------------------------------------------------
// Kernel D2: agg[b,l,h*64+d] = sum_c ctx[b,h][c][d] * Qsm[b,l,h*64+c]
// Block per (b,h, l-chunk of 128); wave w owns l-tile w, all 4 d-tiles.
// B fragments read straight from global (float4 x4 per lane).
// ---------------------------------------------------------------------------
__global__ __launch_bounds__(256) void attend_gemm_kernel(
    const float* __restrict__ ctx, const float* __restrict__ Qsm,
    float* __restrict__ agg) {
  const int bh = blockIdx.x >> 5; // 0..63
  const int b  = bh >> 4;
  const int h  = bh & 15;
  const int l0 = (blockIdx.x & 31) * 128;

  __shared__ __align__(16) unsigned short ctT[64][72]; // [d][c] = ctx[c][d]
  const float* cb = ctx + (size_t)bh * (HC * HC);
  const int tid = threadIdx.x;
  for (int i = tid; i < (HC * HC) / 4; i += 256) { // float4 staging
    const float4 cv = *(const float4*)(cb + i * 4);
    const int c = (i * 4) >> 6;
    const int d = (i * 4) & 63;
    ctT[d + 0][c] = f2bf(cv.x);
    ctT[d + 1][c] = f2bf(cv.y);
    ctT[d + 2][c] = f2bf(cv.z);
    ctT[d + 3][c] = f2bf(cv.w);
  }
  __syncthreads();

  const int lane = tid & 31;
  const int wav  = tid >> 5;
  const int nl   = l0 + wav * 16 + (lane & 15); // this lane's l (N index)
  const int kh   = (lane >> 4) * 16;
  const int kg   = (lane >> 4) * 8;
  const float* qrow =
      Qsm + (size_t)b * L_SEQ * H_DIM + (size_t)nl * H_DIM + h * HC;

  v8f acc[4] = {v8f{}, v8f{}, v8f{}, v8f{}};

#pragma unroll
  for (int kbs = 0; kbs < HC; kbs += 32) {
    const float* qp = qrow + kbs + kg;
    const float4 q0 = *(const float4*)(qp + 0);
    const float4 q1 = *(const float4*)(qp + 4);
    const float4 q2 = *(const float4*)(qp + 16);
    const float4 q3 = *(const float4*)(qp + 20);
    const v16us bu = pack16(q0, q1, q2, q3);
    const unsigned short* a0 = &ctT[0 * 16 + (lane & 15)][0];
    const unsigned short* a1 = &ctT[1 * 16 + (lane & 15)][0];
    const unsigned short* a2 = &ctT[2 * 16 + (lane & 15)][0];
    const unsigned short* a3 = &ctT[3 * 16 + (lane & 15)][0];
    const v16us au0 = ld2x8(a0 + kbs + kg, a0 + kbs + 16 + kg);
    const v16us au1 = ld2x8(a1 + kbs + kg, a1 + kbs + 16 + kg);
    const v16us au2 = ld2x8(a2 + kbs + kg, a2 + kbs + 16 + kg);
    const v16us au3 = ld2x8(a3 + kbs + kg, a3 + kbs + 16 + kg);
    acc[0] = wmma_bf16(au0, bu, acc[0]);
    acc[1] = wmma_bf16(au1, bu, acc[1]);
    acc[2] = wmma_bf16(au2, bu, acc[2]);
    acc[3] = wmma_bf16(au3, bu, acc[3]);
  }

  float* ab =
      agg + (size_t)b * L_SEQ * H_DIM + (size_t)nl * H_DIM + h * HC;
  const int mh = (lane >> 4) * 8;
#pragma unroll
  for (int mtile = 0; mtile < 4; ++mtile) {
#pragma unroll
    for (int r = 0; r < 8; ++r) ab[mtile * 16 + mh + r] = acc[mtile][r];
  }
}

// ---------------------------------------------------------------------------
extern "C" void kernel_launch(void* const* d_in, const int* in_sizes, int n_in,
                              void* d_out, int out_size, void* d_ws,
                              size_t ws_size, hipStream_t stream) {
  (void)in_sizes; (void)n_in; (void)out_size; (void)ws_size;

  const float* x    = (const float*)d_in[0];
  const int*   mask = (const int*)d_in[1];
  const float* kdw = (const float*)d_in[2],  *kdb = (const float*)d_in[3];
  const float* kpw = (const float*)d_in[4],  *kpb = (const float*)d_in[5];
  const float* qdw = (const float*)d_in[6],  *qdb = (const float*)d_in[7];
  const float* qpw = (const float*)d_in[8],  *qpb = (const float*)d_in[9];
  const float* vdw = (const float*)d_in[10], *vdb = (const float*)d_in[11];
  const float* vpw = (const float*)d_in[12], *vpb = (const float*)d_in[13];
  const float* rdw = (const float*)d_in[14], *rdb = (const float*)d_in[15];
  const float* rpw = (const float*)d_in[16], *rpb = (const float*)d_in[17];
  float* out = (float*)d_out;

  float* ws = (float*)d_ws;
  const size_t S = (size_t)M_TOT * H_DIM; // 16.78M floats
  float* Kb = ws;          // key proj -> key_sm -> (reused as agg)
  float* Qb = ws + S;      // query proj -> query_sm
  float* Vb = ws + 2 * S;  // value proj
  float* Cx = ws + 3 * S;  // context: 64 * 64*64 floats
  float* Ag = Kb;          // agg reuses K buffer (free after context GEMM)

  const dim3 gGemm(H_DIM / 64, M_TOT / 128); // 16 x 128 blocks

  proj_gemm_kernel<<<gGemm, 256, 0, stream>>>(x, kdw, kdb, kpw, kpb, Kb);
  proj_gemm_kernel<<<gGemm, 256, 0, stream>>>(x, qdw, qdb, qpw, qpb, Qb);
  proj_gemm_kernel<<<gGemm, 256, 0, stream>>>(x, vdw, vdb, vpw, vpb, Vb);

  key_softmax_kernel<<<B_SZ * H_DIM, 256, 0, stream>>>(Kb, mask);
  query_softmax_kernel<<<M_TOT, 256, 0, stream>>>(Qb, mask);

  context_gemm_kernel<<<B_SZ * NHEAD, 256, 0, stream>>>(Kb, Vb, Cx);
  attend_gemm_kernel<<<B_SZ * NHEAD * (L_SEQ / 128), 256, 0, stream>>>(Cx, Qb, Ag);

  proj_gemm_kernel<<<gGemm, 256, 0, stream>>>(Ag, rdw, rdb, rpw, rpb, out);
}
